// WindowAttention_47390669144727
// MI455X (gfx1250) — matrix operand order repaired and checked by reading, not compiled, not run on hardware
//
#include <hip/hip_runtime.h>
#include <hip/hip_bf16.h>

typedef __attribute__((ext_vector_type(16))) _Float16 v16h;
typedef __attribute__((ext_vector_type(8)))  _Float16 v8h;
typedef __attribute__((ext_vector_type(8)))  float    v8f;
typedef __attribute__((ext_vector_type(4)))  int      v4i;

#define DIM   256
#define HEADS 8
#define HD    32
#define NTOK  64
#define NWIN  1024
#define NBLK  4096
#define WELEM (DIM * DIM)          // elements per weight matrix

#if __has_builtin(__builtin_amdgcn_global_load_async_to_lds_b128)
#define HAVE_ASYNC_LDS 1
#else
#define HAVE_ASYNC_LDS 0
#endif

// typed, address-space-qualified pointers for the async builtin
typedef __attribute__((address_space(1))) v4i g_v4i;   // global b128
typedef __attribute__((address_space(3))) v4i l_v4i;   // LDS b128

// ---- WMMA fragment loaders (contiguous k = 16h + s order, same for A and B) ----

// A-matrix 16x32 f16: lane (0-15) row M=lane holds K 0..15 contiguous; lanes 16-31
// hold K 16..31.  Caller passes pointer already offset by the K group.
__device__ __forceinline__ v16h fragA_lds(const _Float16* p) {
  v8h lo = *(const v8h*)p;
  v8h hi = *(const v8h*)(p + 8);
  v16h r;
  #pragma unroll
  for (int i = 0; i < 8; ++i) { r[i] = lo[i]; r[i + 8] = hi[i]; }
  return r;
}

// B-matrix 32x16 f16: lane (0-15) col N=lane holds K 0..15 contiguous; lanes 16-31
// hold K 16..31.  Caller passes pointer already offset by the K group.
__device__ __forceinline__ v16h fragB_lds(const _Float16* p) {
  v8h lo = *(const v8h*)p;
  v8h hi = *(const v8h*)(p + 8);
  v16h r;
  #pragma unroll
  for (int i = 0; i < 8; ++i) { r[i] = lo[i]; r[i + 8] = hi[i]; }
  return r;
}

// B fragment from f32 weights in global memory (fallback path, inline convert).
__device__ __forceinline__ v16h fragB_w32(const float* p) {
  v16h r;
  #pragma unroll
  for (int i = 0; i < 8; ++i) { r[i] = (_Float16)p[i]; r[i + 8] = (_Float16)p[i + 8]; }
  return r;
}

__device__ __forceinline__ v8f wmma16(v16h a, v16h b, v8f c) {
  return __builtin_amdgcn_wmma_f32_16x16x32_f16(false, a, false, b, (short)0, c,
                                                false, false);
}

// ---- prologue: one-time f32 -> f16 weight conversion into d_ws ----
__global__ __launch_bounds__(256)
void cvt_weights_kernel(const float* __restrict__ wq, const float* __restrict__ wk,
                        const float* __restrict__ wv, const float* __restrict__ wp,
                        _Float16* __restrict__ o) {
  int i = blockIdx.x * 256 + threadIdx.x;      // grid 1024 * 256 = 4*65536
  int m = i >> 16, r = i & (WELEM - 1);
  const float* s = (m == 0) ? wq : (m == 1) ? wk : (m == 2) ? wv : wp;
  o[i] = (_Float16)s[r];
}

template <bool W16>
__global__ __launch_bounds__(256, 1)
void winattn_kernel(const float* __restrict__ x, const float* __restrict__ mask,
                    const float* __restrict__ mw,
                    const float* __restrict__ wq, const float* __restrict__ bq,
                    const float* __restrict__ wk, const float* __restrict__ bk,
                    const float* __restrict__ wv, const float* __restrict__ bv,
                    const float* __restrict__ wp, const float* __restrict__ bp,
                    const _Float16* __restrict__ wh,
                    float* __restrict__ out, float* __restrict__ out_mw)
{
  __shared__ __align__(16) _Float16 sQ [NTOK * DIM];      // 32 KB (scale folded)
  __shared__ __align__(16) _Float16 sK [NTOK * DIM];      // 32 KB
  __shared__ __align__(16) _Float16 sVT[DIM * NTOK];      // 32 KB, transposed V
  __shared__ __align__(16) _Float16 sXbuf[2 * NTOK * DIM];// 64 KB, multi-use
  __shared__ __align__(16) float smask[NTOK * NTOK];      // 16 KB mask+keymask
  __shared__ float skm[NTOK];
  __shared__ float smw[NTOK];

  _Float16* sXn = sXbuf;                // normalized x (f16), dead after QKV
  _Float16* sXr = sXbuf + NTOK * DIM;   // raw x (f16),        dead after QKV
  _Float16* sO  = sXbuf;                // attention output, reuses sXn region
  _Float16* sP  = sXbuf + NTOK * DIM;   // per-wave softmax staging (16x64, stride 72)

  const int tid  = threadIdx.x;
  const int lane = tid & 31;
  const int wid  = tid >> 5;
  const int l15  = lane & 15;
  const int hi8  = (lane >> 4) * 8;     // accumulator row offset (D rows 8h..8h+7)
  const int kh16 = (lane >> 4) * 16;    // B-frag K group offset
  const int b    = blockIdx.x;
  const int wi   = b & (NWIN - 1);      // window index for shifted mask

  // ---------- async global->LDS copy of window mask (ASYNCcnt-tracked) ----------
#if HAVE_ASYNC_LDS
  {
    const float* g = mask + (size_t)wi * NTOK * NTOK + tid * 4;
    float* l = smask + tid * 4;
    #pragma unroll
    for (int it = 0; it < 4; ++it)
      __builtin_amdgcn_global_load_async_to_lds_b128(
          (g_v4i*)(uintptr_t)(g + it * 1024),
          (l_v4i*)(uint32_t)(uintptr_t)(l + it * 1024), 0, 0);
  }
#else
  for (int i = tid; i < NTOK * NTOK; i += 256)
    smask[i] = mask[(size_t)wi * NTOK * NTOK + i];
#endif

  // ---------- Phase A: load x, L2-normalize rows, stage token mask ----------
  {
    const int r  = tid >> 2;            // 4 threads per token row
    const int cq = (tid & 3) * 64;
    const float* xr = x + ((size_t)b * NTOK + r) * DIM + cq;
    _Float16* rr = sXr + r * DIM + cq;
    float ss = 0.f;
    #pragma unroll 8
    for (int c = 0; c < 64; ++c) { float v = xr[c]; rr[c] = (_Float16)v; ss += v * v; }
    ss += __shfl_xor(ss, 1, 32);
    ss += __shfl_xor(ss, 2, 32);
    float scl = 1.f / fmaxf(sqrtf(ss), 1e-12f);
    _Float16* rn = sXn + r * DIM + cq;
    #pragma unroll 8
    for (int c = 0; c < 64; ++c) rn[c] = (_Float16)((float)rr[c] * scl);
  }
  if (tid < NTOK) {
    float v = mw[(size_t)b * NTOK + tid];
    smw[tid] = v;
    skm[tid] = (v == 0.f) ? -100.f : 0.f;
  }
#if HAVE_ASYNC_LDS
#if __has_builtin(__builtin_amdgcn_s_wait_asynccnt)
  __builtin_amdgcn_s_wait_asynccnt(0);
#else
  asm volatile("s_wait_asynccnt 0x0" ::: "memory");
#endif
#endif
  __syncthreads();
  for (int i = tid; i < NTOK * NTOK; i += 256) smask[i] += skm[i & 63];

  // ---------- Phase B: QKV projections, 192 tile-jobs over 8 waves ----------
  const float* Wmat[3] = { wq, wk, wv };
  const float* Bvec[3] = { bq, bk, bv };
  for (int j = 0; j < 24; ++j) {
    int job = wid * 24 + j;
    int mat = job >> 6;                 // 0=Q 1=K 2=V
    int rem = job & 63;
    int mi = rem >> 4, ni = rem & 15;
    const _Float16* Asrc = (mat < 2) ? sXn : sXr;
    float bval = Bvec[mat][ni * 16 + l15];
    v8f acc;
    #pragma unroll
    for (int i = 0; i < 8; ++i) acc[i] = bval;
    #pragma unroll
    for (int kk = 0; kk < 8; ++kk) {
      v16h a = fragA_lds(Asrc + (mi * 16 + l15) * DIM + kk * 32 + kh16);
      v16h bb;
      if constexpr (W16)
        bb = *(const v16h*)(wh + (size_t)mat * WELEM +
                            (size_t)(ni * 16 + l15) * DIM + kk * 32 + kh16);
      else
        bb = fragB_w32(Wmat[mat] + (size_t)(ni * 16 + l15) * DIM + kk * 32 + kh16);
      acc = wmma16(a, bb, acc);
    }
    int col = ni * 16 + l15;
    if (mat == 2) {                     // V stored transposed: sVT[d][m]
      #pragma unroll
      for (int r2 = 0; r2 < 8; ++r2)
        sVT[col * NTOK + mi * 16 + hi8 + r2] = (_Float16)acc[r2];
    } else {
      _Float16* dst = mat ? sK : sQ;
      float s2 = mat ? 1.f : 0.17677669529663687f;   // 1/sqrt(32) folded into Q
      #pragma unroll
      for (int r2 = 0; r2 < 8; ++r2)
        dst[(mi * 16 + hi8 + r2) * DIM + col] = (_Float16)(acc[r2] * s2);
    }
  }
  __syncthreads();

  // ---------- Phase C: attention, wave w owns head w ----------
  {
    const int d0 = wid * HD;
    _Float16* sPw = sP + wid * (16 * 72);
    for (int mi = 0; mi < 4; ++mi) {
      // scores S = q_h @ k_h^T   (M=16 rows, N=64 cols, K=32 = head dim)
      v16h aq = fragA_lds(sQ + (mi * 16 + l15) * DIM + d0 + kh16);
      v8f s0 = {}, s1 = {}, s2 = {}, s3 = {};
      s0 = wmma16(aq, fragB_lds(sK + ( 0 + l15) * DIM + d0 + kh16), s0);
      s1 = wmma16(aq, fragB_lds(sK + (16 + l15) * DIM + d0 + kh16), s1);
      s2 = wmma16(aq, fragB_lds(sK + (32 + l15) * DIM + d0 + kh16), s2);
      s3 = wmma16(aq, fragB_lds(sK + (48 + l15) * DIM + d0 + kh16), s3);
      // mask add + exact softmax (rows are per-VGPR in C layout; reduce 16 lanes)
      #pragma unroll
      for (int r = 0; r < 8; ++r) {
        const float* mrow = smask + (mi * 16 + hi8 + r) * NTOK + l15;
        float t0 = s0[r] + mrow[0],  t1 = s1[r] + mrow[16];
        float t2 = s2[r] + mrow[32], t3 = s3[r] + mrow[48];
        float mx = fmaxf(fmaxf(t0, t1), fmaxf(t2, t3));
        mx = fmaxf(mx, __shfl_xor(mx, 1, 32));
        mx = fmaxf(mx, __shfl_xor(mx, 2, 32));
        mx = fmaxf(mx, __shfl_xor(mx, 4, 32));
        mx = fmaxf(mx, __shfl_xor(mx, 8, 32));
        float e0 = __expf(t0 - mx), e1 = __expf(t1 - mx);
        float e2 = __expf(t2 - mx), e3 = __expf(t3 - mx);
        float sm = e0 + e1 + e2 + e3;
        sm += __shfl_xor(sm, 1, 32);
        sm += __shfl_xor(sm, 2, 32);
        sm += __shfl_xor(sm, 4, 32);
        sm += __shfl_xor(sm, 8, 32);
        float inv = 1.f / sm;
        _Float16* pr = sPw + (hi8 + r) * 72 + l15;
        pr[0]  = (_Float16)(e0 * inv);
        pr[16] = (_Float16)(e1 * inv);
        pr[32] = (_Float16)(e2 * inv);
        pr[48] = (_Float16)(e3 * inv);
      }
      // O row-block = P(16x64) @ v_h(64x32)
      v8f o0 = {}, o1 = {};
      #pragma unroll
      for (int kk = 0; kk < 2; ++kk) {
        v16h ap = fragA_lds(sPw + l15 * 72 + kk * 32 + kh16);
        o0 = wmma16(ap, fragB_lds(sVT + (d0      + l15) * NTOK + kk * 32 + kh16), o0);
        o1 = wmma16(ap, fragB_lds(sVT + (d0 + 16 + l15) * NTOK + kk * 32 + kh16), o1);
      }
      #pragma unroll
      for (int r = 0; r < 8; ++r) {
        sO[(mi * 16 + hi8 + r) * DIM + d0      + l15] = (_Float16)o0[r];
        sO[(mi * 16 + hi8 + r) * DIM + d0 + 16 + l15] = (_Float16)o1[r];
      }
    }
  }
  __syncthreads();

  // ---------- Phase D: output projection, 64 tile-jobs over 8 waves ----------
  for (int j = 0; j < 8; ++j) {
    int job = wid * 8 + j;
    int mi = job >> 4, ni = job & 15;
    float bval = bp[ni * 16 + l15];
    v8f acc;
    #pragma unroll
    for (int i = 0; i < 8; ++i) acc[i] = bval;
    #pragma unroll
    for (int kk = 0; kk < 8; ++kk) {
      v16h a = fragA_lds(sO + (mi * 16 + l15) * DIM + kk * 32 + kh16);
      v16h bb;
      if constexpr (W16)
        bb = *(const v16h*)(wh + (size_t)3 * WELEM +
                            (size_t)(ni * 16 + l15) * DIM + kk * 32 + kh16);
      else
        bb = fragB_w32(wp + (size_t)(ni * 16 + l15) * DIM + kk * 32 + kh16);
      acc = wmma16(a, bb, acc);
    }
    #pragma unroll
    for (int r = 0; r < 8; ++r)
      out[((size_t)b * NTOK + mi * 16 + hi8 + r) * DIM + ni * 16 + l15] = acc[r];
  }

  // ---------- Phase E: new mask_windows = clip(sum_n mw, 0, 1), tiled ----------
  if (tid < NTOK) {
    float sm = 0.f;
    #pragma unroll 8
    for (int i = 0; i < NTOK; ++i) sm += smw[i];
    out_mw[(size_t)b * NTOK + tid] = fminf(sm, 1.f);
  }
}

extern "C" void kernel_launch(void* const* d_in, const int* in_sizes, int n_in,
                              void* d_out, int out_size, void* d_ws, size_t ws_size,
                              hipStream_t stream) {
  const float* x    = (const float*)d_in[0];
  const float* mask = (const float*)d_in[1];
  const float* mw   = (const float*)d_in[2];
  const float* wq   = (const float*)d_in[3];
  const float* bq   = (const float*)d_in[4];
  const float* wk   = (const float*)d_in[5];
  const float* bk   = (const float*)d_in[6];
  const float* wv   = (const float*)d_in[7];
  const float* bv   = (const float*)d_in[8];
  const float* wp   = (const float*)d_in[9];
  const float* bp   = (const float*)d_in[10];
  float* out    = (float*)d_out;
  float* out_mw = out + (size_t)NBLK * NTOK * DIM;   // tuple output: concat flat

  const size_t wbytes = (size_t)4 * WELEM * sizeof(_Float16);   // 512 KB
  if (ws_size >= wbytes) {
    _Float16* wh = (_Float16*)d_ws;
    cvt_weights_kernel<<<(4 * WELEM) / 256, 256, 0, stream>>>(wq, wk, wv, wp, wh);
    winattn_kernel<true><<<NBLK, 256, 0, stream>>>(x, mask, mw, wq, bq, wk, bk,
                                                   wv, bv, wp, bp, wh, out, out_mw);
  } else {
    winattn_kernel<false><<<NBLK, 256, 0, stream>>>(x, mask, mw, wq, bq, wk, bk,
                                                    wv, bv, wp, bp, nullptr,
                                                    out, out_mw);
  }
}
